// BiDirectionalMinGRU_75531294867630
// MI455X (gfx1250) — matrix operand order repaired and checked by reading, not compiled, not run on hardware
//
#include <hip/hip_runtime.h>
#include <hip/hip_bf16.h>
#include <math.h>

// Problem sizes (match reference)
#define BB      16
#define LL      8192
#define HD      256             // H
#define TE_DIM  8
#define OUT_DIM (2*HD + TE_DIM) // 520
#define KPAD    544             // OUT_DIM padded to multiple of 32 (17*32)
#define GHID    128             // max(32, H//2)
#define BL      (BB*LL)         // 131072 positions
#define NCH     32              // scan chunks per sequence
#define CH      (LL/NCH)        // 256 steps per chunk

typedef __attribute__((ext_vector_type(16))) __bf16 v16bf;
typedef __attribute__((ext_vector_type(8)))  __bf16 v8bf;
typedef __attribute__((ext_vector_type(8)))  float  v8f;

// ---------- math helpers ----------
__device__ __forceinline__ float softplus_f(float x) {
  return fmaxf(x, 0.f) + log1pf(__expf(-fabsf(x)));   // stable softplus
}
__device__ __forceinline__ float log_g_f(float x) {
  return (x >= 0.f) ? __logf(x + 0.5f) : (5.f - softplus_f(-x));
}
__device__ __forceinline__ float gelu_tanh(float x) {
  const float c = 0.7978845608028654f;
  return 0.5f * x * (1.f + tanhf(c * (x + 0.044715f * x * x * x)));
}
__device__ __forceinline__ float logaddexp_f(float x1, float x2) {
  return fmaxf(x1, x2) + log1pf(__expf(-fabsf(x1 - x2)));
}

// ---------- K0a: convert 4x (256x256) f32 gate weights to bf16, transposed ----------
__global__ void convert_weights_kernel(const float* __restrict__ wzf, const float* __restrict__ whf,
                                       const float* __restrict__ wzb, const float* __restrict__ whb,
                                       __bf16* __restrict__ ozf, __bf16* __restrict__ ohf,
                                       __bf16* __restrict__ ozb, __bf16* __restrict__ ohb) {
  int idx = blockIdx.x * blockDim.x + threadIdx.x;   // 4*65536 threads
  int mat = idx >> 16, e = idx & 65535;
  int k = e >> 8, n = e & 255;
  const float* src = (mat == 0) ? wzf : (mat == 1) ? whf : (mat == 2) ? wzb : whb;
  __bf16*      dst = (mat == 0) ? ozf : (mat == 1) ? ohf : (mat == 2) ? ozb : ohb;
  dst[n * HD + k] = (__bf16)src[k * HD + n];
}

// ---------- K0b: convert head W1 (520x128) -> bf16 transposed + K-padded (128x544) ----------
__global__ void convert_w1_kernel(const float* __restrict__ w1, __bf16* __restrict__ w1t) {
  int idx = blockIdx.x * blockDim.x + threadIdx.x;   // GHID*KPAD = 69632 threads
  if (idx >= GHID * KPAD) return;
  int n = idx / KPAD, k = idx % KPAD;
  w1t[idx] = (k < OUT_DIM) ? (__bf16)w1[k * GHID + n] : (__bf16)0.f;
}

// ---------- K1: time encoding + 10->256 input projections (both directions), bf16 out ----------
__global__ void __launch_bounds__(256)
input_proj_kernel(const float* __restrict__ x, const float* __restrict__ t,
                  const float* __restrict__ mask,
                  const float* __restrict__ te_w1, const float* __restrict__ te_b1,
                  const float* __restrict__ te_w2, const float* __restrict__ te_b2,
                  const float* __restrict__ fproj_w, const float* __restrict__ fproj_b,
                  const float* __restrict__ bproj_w, const float* __restrict__ bproj_b,
                  const float* __restrict__ time_scale,
                  float* __restrict__ t_enc_out,
                  __bf16* __restrict__ u_f, __bf16* __restrict__ u_b) {
  int pos = blockIdx.x;
  int j   = threadIdx.x;
  int b   = pos >> 13;                               // / LL
  float ts = (t[pos] - t[b * LL]) / time_scale[0];
  float h1[TE_DIM], te[TE_DIM];
  #pragma unroll
  for (int i = 0; i < TE_DIM; ++i) h1[i] = fmaxf(ts * te_w1[i] + te_b1[i], 0.f);
  #pragma unroll
  for (int e = 0; e < TE_DIM; ++e) {
    float a = te_b2[e];
    #pragma unroll
    for (int i = 0; i < TE_DIM; ++i) a += h1[i] * te_w2[i * TE_DIM + e];
    te[e] = a;
  }
  float m = mask[pos];
  float rnn[10];
  rnn[0] = x[(size_t)pos * 2 + 0] * m;
  rnn[1] = x[(size_t)pos * 2 + 1] * m;
  #pragma unroll
  for (int e = 0; e < TE_DIM; ++e) rnn[2 + e] = te[e];
  float uf = fproj_b[j], ub = bproj_b[j];
  #pragma unroll
  for (int i = 0; i < 10; ++i) {
    uf += rnn[i] * fproj_w[i * HD + j];
    ub += rnn[i] * bproj_w[i * HD + j];
  }
  size_t base = (size_t)pos * HD;
  u_f[base + j] = (__bf16)uf;
  u_b[base + j] = (__bf16)ub;
  if (j < TE_DIM) t_enc_out[(size_t)pos * TE_DIM + j] = te[j];
}

// A-fragment loader (contiguous K layout): lane half selects K {0-15}/{16-31}
__device__ __forceinline__ v16bf load_a_frag(const __bf16* p) {
  v8bf lo = *(const v8bf*)p;
  v8bf hi = *(const v8bf*)(p + 8);
  v16bf r;
  #pragma unroll
  for (int i = 0; i < 8; ++i) { r[i] = lo[i]; r[i + 8] = hi[i]; }
  return r;
}

// ---------- K2: gate GEMMs with WMMA bf16 -> f32, fused log-space epilogue ----------
__global__ void __launch_bounds__(256)
gate_gemm_kernel(const __bf16* __restrict__ U,
                 const __bf16* __restrict__ WzT, const __bf16* __restrict__ WhT,
                 const float* __restrict__ bz, const float* __restrict__ bh,
                 float* __restrict__ logA, float* __restrict__ logB) {
  const int m0   = blockIdx.x * 16;
  const int lane = threadIdx.x & 31;
  const int wave = threadIdx.x >> 5;      // 8 waves cover N = 0..255
  const int r    = lane & 15;
  const int hi   = lane >> 4;
  const int n0   = wave * 32;

  v8f cz0 = {}; v8f cz1 = {}; v8f ch0 = {}; v8f ch1 = {};

  const __bf16* arow = U   + (size_t)(m0 + r)      * HD + hi * 16;
  const __bf16* bz0p = WzT + (size_t)(n0 + r)      * HD + hi * 16;
  const __bf16* bz1p = WzT + (size_t)(n0 + 16 + r) * HD + hi * 16;
  const __bf16* bh0p = WhT + (size_t)(n0 + r)      * HD + hi * 16;
  const __bf16* bh1p = WhT + (size_t)(n0 + 16 + r) * HD + hi * 16;

  __builtin_prefetch(arow + 16 * HD, 0, 1);         // next block's A strip

  #pragma unroll
  for (int k0 = 0; k0 < HD; k0 += 32) {
    v16bf a  = load_a_frag(arow + k0);
    v16bf b0 = *(const v16bf*)(bz0p + k0);
    v16bf b1 = *(const v16bf*)(bz1p + k0);
    v16bf b2 = *(const v16bf*)(bh0p + k0);
    v16bf b3 = *(const v16bf*)(bh1p + k0);
    cz0 = __builtin_amdgcn_wmma_f32_16x16x32_bf16(false, a, false, b0, (short)0, cz0, false, false);
    cz1 = __builtin_amdgcn_wmma_f32_16x16x32_bf16(false, a, false, b1, (short)0, cz1, false, false);
    ch0 = __builtin_amdgcn_wmma_f32_16x16x32_bf16(false, a, false, b2, (short)0, ch0, false, false);
    ch1 = __builtin_amdgcn_wmma_f32_16x16x32_bf16(false, a, false, b3, (short)0, ch1, false, false);
  }

  const int nA = n0 + r;
  const int nB = n0 + 16 + r;
  const float bzA = bz[nA], bzB = bz[nB], bhA = bh[nA], bhB = bh[nB];
  #pragma unroll
  for (int v = 0; v < 8; ++v) {
    const int m = m0 + v + hi * 8;
    const size_t oA = (size_t)m * HD + nA;
    const size_t oB = (size_t)m * HD + nB;
    float kkA = cz0[v] + bzA, ggA = ch0[v] + bhA;
    float kkB = cz1[v] + bzB, ggB = ch1[v] + bhB;
    logA[oA] = -softplus_f(kkA);
    logB[oA] = -softplus_f(-kkA) + log_g_f(ggA);
    logA[oB] = -softplus_f(kkB);
    logB[oB] = -softplus_f(-kkB) + log_g_f(ggB);
  }
}

// ---------- K3: chunked parallel log-space scan (3 phases) ----------
// Recurrence logh <- logaddexp(a + logh, b) composes as (a1,b1)o(a2,b2) = (a1+a2, lae(a2+b1, b2)).
// Phase A: per-chunk composition.  grid (B*H*NCH/256, 2dirs)
__global__ void __launch_bounds__(256)
scan_phaseA_kernel(const float* __restrict__ laf, const float* __restrict__ lbf,
                   const float* __restrict__ lab, const float* __restrict__ lbb,
                   float* __restrict__ a_sum, float* __restrict__ b_sum) {
  int idx = blockIdx.x * blockDim.x + threadIdx.x;   // (b, c, ch)
  int dir = blockIdx.y;
  int ch = idx & 255, c = (idx >> 8) & (NCH - 1), b = idx >> 13;
  const float* la = dir ? lab : laf;
  const float* lb = dir ? lbb : lbf;
  float a = 0.f, bv = -INFINITY;
  for (int s = c * CH; s < (c + 1) * CH; ++s) {
    int tt = dir ? (LL - 1 - s) : s;
    size_t o = ((size_t)b * LL + tt) * HD + ch;
    float va = la[o], vb = lb[o];
    bv = logaddexp_f(bv + va, vb);
    a += va;
  }
  size_t so = (size_t)dir * (BB * HD * NCH) + idx;
  a_sum[so] = a;
  b_sum[so] = bv;
}

// Phase B: exclusive scan over chunk summaries per sequence.  grid (B*H/256, 2dirs)
__global__ void __launch_bounds__(256)
scan_phaseB_kernel(const float* __restrict__ a_sum, const float* __restrict__ b_sum,
                   float* __restrict__ prefix) {
  int idx = blockIdx.x * blockDim.x + threadIdx.x;   // (b, ch)
  int dir = blockIdx.y;
  int ch = idx & 255, b = idx >> 8;
  size_t dbase = (size_t)dir * (BB * HD * NCH);
  float p = -0.69314718055994531f;                   // log(0.5) = log_g(0)
  #pragma unroll
  for (int c = 0; c < NCH; ++c) {
    size_t so = dbase + ((size_t)b * NCH + c) * HD + ch;
    prefix[so] = p;
    p = logaddexp_f(a_sum[so] + p, b_sum[so]);
  }
}

// Phase C: replay chunk from its prefix, write h = exp(logh).  grid (B*H*NCH/256, 2dirs)
__global__ void __launch_bounds__(256)
scan_phaseC_kernel(const float* __restrict__ laf, const float* __restrict__ lbf,
                   const float* __restrict__ lab, const float* __restrict__ lbb,
                   const float* __restrict__ prefix,
                   float* __restrict__ hf, float* __restrict__ hb) {
  int idx = blockIdx.x * blockDim.x + threadIdx.x;
  int dir = blockIdx.y;
  int ch = idx & 255, c = (idx >> 8) & (NCH - 1), b = idx >> 13;
  const float* la = dir ? lab : laf;
  const float* lb = dir ? lbb : lbf;
  float*       h  = dir ? hb  : hf;
  float logh = prefix[(size_t)dir * (BB * HD * NCH) + idx];
  for (int s = c * CH; s < (c + 1) * CH; ++s) {
    int tt = dir ? (LL - 1 - s) : s;
    size_t o = ((size_t)b * LL + tt) * HD + ch;
    logh = logaddexp_f(la[o] + logh, lb[o]);
    h[o] = __expf(logh);
  }
}

// ---------- K4a: concat + LayerNorm -> bf16 activations (K-padded to 544) ----------
__global__ void __launch_bounds__(128)
ln_store_kernel(const float* __restrict__ hf, const float* __restrict__ hb,
                const float* __restrict__ tenc,
                const float* __restrict__ ln_g, const float* __restrict__ ln_b,
                __bf16* __restrict__ act) {
  __shared__ float s_v[OUT_DIM];
  __shared__ float s_red[8];
  const int pos = blockIdx.x;
  const int tid = threadIdx.x;
  const size_t hbase = (size_t)pos * HD;

  float sum = 0.f, sq = 0.f;
  for (int i = tid; i < OUT_DIM; i += 128) {
    float v = (i < HD)     ? hf[hbase + i]
            : (i < 2 * HD) ? hb[hbase + (i - HD)]
                           : tenc[(size_t)pos * TE_DIM + (i - 2 * HD)];
    s_v[i] = v; sum += v; sq += v * v;
  }
  #pragma unroll
  for (int off = 16; off > 0; off >>= 1) {
    sum += __shfl_xor(sum, off, 32);
    sq  += __shfl_xor(sq,  off, 32);
  }
  const int wv = tid >> 5;
  if ((tid & 31) == 0) { s_red[wv] = sum; s_red[4 + wv] = sq; }
  __syncthreads();
  float tsum = s_red[0] + s_red[1] + s_red[2] + s_red[3];
  float tsq  = s_red[4] + s_red[5] + s_red[6] + s_red[7];
  float mu  = tsum / (float)OUT_DIM;
  float var = tsq / (float)OUT_DIM - mu * mu;
  float rs  = rsqrtf(var + 1e-5f);
  size_t abase = (size_t)pos * KPAD;
  for (int i = tid; i < OUT_DIM; i += 128)
    act[abase + i] = (__bf16)((s_v[i] - mu) * rs * ln_g[i] + ln_b[i]);
  if (tid < KPAD - OUT_DIM) act[abase + OUT_DIM + tid] = (__bf16)0.f;   // zero K-pad
}

// ---------- K4b: head GEMM (BLx544 @ 544x128) via WMMA + fused gelu.w2 reduction ----------
__global__ void __launch_bounds__(256)
head_gemm_kernel(const __bf16* __restrict__ act, const __bf16* __restrict__ w1t,
                 const float* __restrict__ b1, const float* __restrict__ w2,
                 const float* __restrict__ b2, float* __restrict__ out) {
  __shared__ float s_part[8 * 16];                   // [wave][row]
  const int m0   = blockIdx.x * 16;
  const int lane = threadIdx.x & 31;
  const int wave = threadIdx.x >> 5;                 // 8 waves cover N = 0..127 (16 each)
  const int r    = lane & 15;
  const int hi   = lane >> 4;
  const int n0   = wave * 16;

  v8f acc = {};
  const __bf16* arow = act + (size_t)(m0 + r) * KPAD + hi * 16;
  const __bf16* brow = w1t + (size_t)(n0 + r) * KPAD + hi * 16;

  __builtin_prefetch(arow + (size_t)16 * KPAD, 0, 1);

  #pragma unroll
  for (int k0 = 0; k0 < KPAD; k0 += 32) {
    v16bf a = load_a_frag(arow + k0);
    v16bf b = *(const v16bf*)(brow + k0);
    acc = __builtin_amdgcn_wmma_f32_16x16x32_bf16(false, a, false, b, (short)0, acc, false, false);
  }

  // epilogue: hidden = gelu(acc + b1[n]); partial = hidden * w2[n]; reduce over N
  const int n = n0 + r;
  const float b1v = b1[n], w2v = w2[n];
  float part[8];
  #pragma unroll
  for (int v = 0; v < 8; ++v) part[v] = gelu_tanh(acc[v] + b1v) * w2v;
  // reduce across the 16-lane half (bits 0..3 only -> stays within half)
  #pragma unroll
  for (int off = 1; off < 16; off <<= 1) {
    #pragma unroll
    for (int v = 0; v < 8; ++v) part[v] += __shfl_xor(part[v], off, 32);
  }
  if (r == 0) {                                      // lane 0 -> rows v; lane 16 -> rows v+8
    #pragma unroll
    for (int v = 0; v < 8; ++v) s_part[wave * 16 + v + hi * 8] = part[v];
  }
  __syncthreads();
  if (threadIdx.x < 16) {
    float total = b2[0];
    #pragma unroll
    for (int w = 0; w < 8; ++w) total += s_part[w * 16 + threadIdx.x];
    out[m0 + threadIdx.x] = total;
  }
}

extern "C" void kernel_launch(void* const* d_in, const int* in_sizes, int n_in,
                              void* d_out, int out_size, void* d_ws, size_t ws_size,
                              hipStream_t stream) {
  const float* x        = (const float*)d_in[0];
  const float* t        = (const float*)d_in[1];
  const float* mask     = (const float*)d_in[2];
  const float* te_w1    = (const float*)d_in[3];
  const float* te_b1    = (const float*)d_in[4];
  const float* te_w2    = (const float*)d_in[5];
  const float* te_b2    = (const float*)d_in[6];
  const float* fproj_w  = (const float*)d_in[7];
  const float* fproj_b  = (const float*)d_in[8];
  const float* bproj_w  = (const float*)d_in[9];
  const float* bproj_b  = (const float*)d_in[10];
  const float* f_wz     = (const float*)d_in[11];
  const float* f_bz     = (const float*)d_in[12];
  const float* f_wh     = (const float*)d_in[13];
  const float* f_bh     = (const float*)d_in[14];
  const float* b_wz     = (const float*)d_in[15];
  const float* b_bz     = (const float*)d_in[16];
  const float* b_wh     = (const float*)d_in[17];
  const float* b_bh     = (const float*)d_in[18];
  const float* ln_g     = (const float*)d_in[19];
  const float* ln_b     = (const float*)d_in[20];
  const float* gh_w1    = (const float*)d_in[21];
  const float* gh_b1    = (const float*)d_in[22];
  const float* gh_w2    = (const float*)d_in[23];
  const float* gh_b2    = (const float*)d_in[24];
  const float* tscale   = (const float*)d_in[25];

  // workspace layout
  char* ws = (char*)d_ws;
  size_t off = 0;
  auto take = [&](size_t bytes) { char* p = ws + off; off += (bytes + 255) & ~(size_t)255; return p; };
  __bf16* wzt_f = (__bf16*)take((size_t)HD * HD * 2);
  __bf16* wht_f = (__bf16*)take((size_t)HD * HD * 2);
  __bf16* wzt_b = (__bf16*)take((size_t)HD * HD * 2);
  __bf16* wht_b = (__bf16*)take((size_t)HD * HD * 2);
  __bf16* w1t   = (__bf16*)take((size_t)GHID * KPAD * 2);
  float*  tenc  = (float*) take((size_t)BL * TE_DIM * 4);
  __bf16* u_f   = (__bf16*)take((size_t)BL * HD * 2);
  __bf16* u_b   = (__bf16*)take((size_t)BL * HD * 2);
  float*  laf   = (float*) take((size_t)BL * HD * 4);
  float*  lbf   = (float*) take((size_t)BL * HD * 4);
  float*  lab   = (float*) take((size_t)BL * HD * 4);
  float*  lbb   = (float*) take((size_t)BL * HD * 4);
  float*  h_f   = (float*) take((size_t)BL * HD * 4);
  float*  h_b   = (float*) take((size_t)BL * HD * 4);
  float*  a_sum = (float*) take((size_t)2 * BB * HD * NCH * 4);
  float*  b_sum = (float*) take((size_t)2 * BB * HD * NCH * 4);
  float*  pfx   = (float*) take((size_t)2 * BB * HD * NCH * 4);
  __bf16* act   = (__bf16*)take((size_t)BL * KPAD * 2);
  (void)ws_size; (void)in_sizes; (void)n_in; (void)out_size;

  convert_weights_kernel<<<(4 * HD * HD) / 256, 256, 0, stream>>>(
      f_wz, f_wh, b_wz, b_wh, wzt_f, wht_f, wzt_b, wht_b);
  convert_w1_kernel<<<(GHID * KPAD + 255) / 256, 256, 0, stream>>>(gh_w1, w1t);

  input_proj_kernel<<<BL, 256, 0, stream>>>(
      x, t, mask, te_w1, te_b1, te_w2, te_b2,
      fproj_w, fproj_b, bproj_w, bproj_b, tscale, tenc, u_f, u_b);

  gate_gemm_kernel<<<BL / 16, 256, 0, stream>>>(u_f, wzt_f, wht_f, f_bz, f_bh, laf, lbf);
  gate_gemm_kernel<<<BL / 16, 256, 0, stream>>>(u_b, wzt_b, wht_b, b_bz, b_bh, lab, lbb);

  scan_phaseA_kernel<<<dim3((BB * HD * NCH) / 256, 2), 256, 0, stream>>>(
      laf, lbf, lab, lbb, a_sum, b_sum);
  scan_phaseB_kernel<<<dim3((BB * HD) / 256, 2), 256, 0, stream>>>(a_sum, b_sum, pfx);
  scan_phaseC_kernel<<<dim3((BB * HD * NCH) / 256, 2), 256, 0, stream>>>(
      laf, lbf, lab, lbb, pfx, h_f, h_b);

  ln_store_kernel<<<BL, 128, 0, stream>>>(h_f, h_b, tenc, ln_g, ln_b, act);
  head_gemm_kernel<<<BL / 16, 256, 0, stream>>>(act, w1t, gh_b1, gh_w2, gh_b2, (float*)d_out);
}
